// LocalWindowAttentionLayer_70660801954254
// MI455X (gfx1250) — matrix operand order repaired and checked by reading, not compiled, not run on hardware
//
#include <hip/hip_runtime.h>
#include <hip/hip_bf16.h>
#include <math.h>

#define B_   4
#define L_   2048
#define D_   512
#define H_   8
#define DK_  64
#define DF_  2048
#define PAD_ 4
#define LP_  (L_ + 2 * PAD_)   // 2056 padded rows per batch
#define M_   (B_ * L_)         // 8192 token rows

typedef _Float16 v8h  __attribute__((ext_vector_type(8)));
typedef _Float16 v16h __attribute__((ext_vector_type(16)));
typedef float    v8f  __attribute__((ext_vector_type(8)));

static __device__ __forceinline__ v8f v8f_zero() {
  v8f z;
#pragma unroll
  for (int i = 0; i < 8; ++i) z[i] = 0.0f;
  return z;
}

// A fragment (16x32 f16): lane<16 holds K {0..7,16..23} of row (lane&15),
// lane>=16 holds K {8..15,24..31}.  p = rowBase + k0 + (lane<16 ? 0 : 8).
static __device__ __forceinline__ v16h ldfragA(const _Float16* p) {
  v8h lo = *(const v8h*)(p);        // 16B aligned
  v8h hi = *(const v8h*)(p + 16);   // +32B
  v16h r;
#pragma unroll
  for (int i = 0; i < 8; ++i) { r[i] = lo[i]; r[i + 8] = hi[i]; }
  return r;
}

// B fragment (32x16 f16): lane = column (lane&15); lane<16 holds K {0..7,16..23},
// lane>=16 holds K {8..15,24..31}.  p = colBase + k0 + (lane<16 ? 0 : 8).
static __device__ __forceinline__ v16h ldfragB(const _Float16* p) {
  v8h lo = *(const v8h*)(p);
  v8h hi = *(const v8h*)(p + 16);
  v16h r;
#pragma unroll
  for (int i = 0; i < 8; ++i) { r[i] = lo[i]; r[i + 8] = hi[i]; }
  return r;
}

struct Frags { v16h a0, a1, b0, b1, b2, b3; };

static __device__ __forceinline__ Frags
load_frags(const _Float16* aRow0, const _Float16* aRow1,
           const _Float16* bRow0, const _Float16* bRow1,
           const _Float16* bRow2, const _Float16* bRow3, int k0) {
  Frags f;
  f.a0 = ldfragA(aRow0 + k0);
  f.a1 = ldfragA(aRow1 + k0);
  f.b0 = ldfragB(bRow0 + k0);
  f.b1 = ldfragB(bRow1 + k0);
  f.b2 = ldfragB(bRow2 + k0);
  f.b3 = ldfragB(bRow3 + k0);
  return f;
}

static __device__ __forceinline__ void mma8(v8f acc[2][4], const Frags& f) {
  acc[0][0] = __builtin_amdgcn_wmma_f32_16x16x32_f16(false, f.a0, false, f.b0, (short)0, acc[0][0], false, false);
  acc[1][0] = __builtin_amdgcn_wmma_f32_16x16x32_f16(false, f.a1, false, f.b0, (short)0, acc[1][0], false, false);
  acc[0][1] = __builtin_amdgcn_wmma_f32_16x16x32_f16(false, f.a0, false, f.b1, (short)0, acc[0][1], false, false);
  acc[1][1] = __builtin_amdgcn_wmma_f32_16x16x32_f16(false, f.a1, false, f.b1, (short)0, acc[1][1], false, false);
  acc[0][2] = __builtin_amdgcn_wmma_f32_16x16x32_f16(false, f.a0, false, f.b2, (short)0, acc[0][2], false, false);
  acc[1][2] = __builtin_amdgcn_wmma_f32_16x16x32_f16(false, f.a1, false, f.b2, (short)0, acc[1][2], false, false);
  acc[0][3] = __builtin_amdgcn_wmma_f32_16x16x32_f16(false, f.a0, false, f.b3, (short)0, acc[0][3], false, false);
  acc[1][3] = __builtin_amdgcn_wmma_f32_16x16x32_f16(false, f.a1, false, f.b3, (short)0, acc[1][3], false, false);
}

// ---------------------------------------------------------------------------
// Generic GEMM: C[M,N] = A[M,K](f16) * Bt[N,K](f16)^T + bias
// mode 0: store f32                (C32)
// mode 1: store f32 into per-batch padded layout (QKV -> (B,LP,D))
// mode 2: exact GELU, store f16    (C16)
// One wave computes a 32x64 tile.  K is a multiple of 64.  The K loop is a
// manual 2-stage ping-pong (loads for step n+1 issued before wmmas of step
// n); "#pragma unroll 1" keeps the compiler from doubling it again, which
// previously created 4 live fragment buffers and spilled to scratch.  The
// last 64-K step is peeled so no conditional reload lives inside the loop.
// ---------------------------------------------------------------------------
__global__ void __launch_bounds__(256, 1)
gemm_wmma_kernel(const _Float16* __restrict__ A,
                 const _Float16* __restrict__ Bt,
                 const float* __restrict__ bias,
                 float* __restrict__ C32,
                 _Float16* __restrict__ C16,
                 int M, int N, int K, int mode) {
  const int lane = threadIdx.x & 31;
  const int w    = (blockIdx.x * blockDim.x + threadIdx.x) >> 5;
  const int nT   = N >> 6;
  const int mi   = w / nT;
  const int ni   = w - mi * nT;
  if (mi * 32 >= M) return;                       // wave-uniform guard
  const int m0 = mi * 32, n0 = ni * 64;
  const int lr = lane & 15;
  const int kShiftA = (lane < 16) ? 0 : 8;
  const int kShiftB = (lane < 16) ? 0 : 8;

  const _Float16* aRow0 = A + (size_t)(m0 + lr) * K + kShiftA;
  const _Float16* aRow1 = aRow0 + (size_t)16 * K;
  const _Float16* bRow0 = Bt + (size_t)(n0 + lr) * K + kShiftB;
  const _Float16* bRow1 = bRow0 + (size_t)16 * K;
  const _Float16* bRow2 = bRow0 + (size_t)32 * K;
  const _Float16* bRow3 = bRow0 + (size_t)48 * K;

  v8f acc[2][4];
#pragma unroll
  for (int t = 0; t < 2; ++t)
#pragma unroll
    for (int j = 0; j < 4; ++j) acc[t][j] = v8f_zero();

  Frags f0 = load_frags(aRow0, aRow1, bRow0, bRow1, bRow2, bRow3, 0);
#pragma unroll 1
  for (int k0 = 0; k0 < K - 64; k0 += 64) {
    Frags f1 = load_frags(aRow0, aRow1, bRow0, bRow1, bRow2, bRow3, k0 + 32);
    __builtin_prefetch((const void*)(aRow0 + k0 + 256), 0, 1);
    __builtin_prefetch((const void*)(bRow0 + k0 + 256), 0, 1);
    mma8(acc, f0);                 // f0 dead here; its regs reused below
    f0 = load_frags(aRow0, aRow1, bRow0, bRow1, bRow2, bRow3, k0 + 64);
    mma8(acc, f1);
  }
  {  // peeled final 64-K step
    Frags f1 = load_frags(aRow0, aRow1, bRow0, bRow1, bRow2, bRow3, K - 32);
    mma8(acc, f0);
    mma8(acc, f1);
  }

  // Epilogue.  C/D layout: VGPR i -> row base+i (lanes 0-15) / base+8+i.
  const int colBase = n0 + lr;
  const int rBase   = m0 + ((lane < 16) ? 0 : 8);
#pragma unroll
  for (int t = 0; t < 2; ++t) {
#pragma unroll
    for (int j = 0; j < 4; ++j) {
      const int c = colBase + 16 * j;
      const float bb = bias[c];
#pragma unroll
      for (int i = 0; i < 8; ++i) {
        const int row = rBase + 16 * t + i;
        float v = acc[t][j][i] + bb;
        if (mode == 2) {
          v = 0.5f * v * (1.0f + erff(v * 0.70710678118654752f));  // exact GELU
          C16[(size_t)row * N + c] = (_Float16)v;
        } else {
          size_t prow = (mode == 1) ? (size_t)(row + PAD_ + 2 * PAD_ * (row >> 11))
                                    : (size_t)row;
          C32[prow * N + c] = v;
        }
      }
    }
  }
}

// ---------------------------------------------------------------------------
__global__ void convert_f32_to_f16(const float* __restrict__ in,
                                   _Float16* __restrict__ out, int n) {
  int t = blockIdx.x * blockDim.x + threadIdx.x;
  if (t < n) out[t] = (_Float16)in[t];
}

// W (K x N, row-major f32) -> Wt (N x K, row-major f16)
__global__ void transpose_to_f16(const float* __restrict__ W,
                                 _Float16* __restrict__ Wt, int K, int N) {
  int t = blockIdx.x * blockDim.x + threadIdx.x;
  if (t >= K * N) return;
  int n = t % N, k = t / N;
  Wt[(size_t)n * K + k] = (_Float16)W[t];
}

// Fill the 8 pad rows of each batch of Qp/Kp/Vp with the bias vectors
// (padded x rows are zero, so projected padded rows equal the bias).
__global__ void pad_init_kernel(float* __restrict__ Qp, float* __restrict__ Kp,
                                float* __restrict__ Vp,
                                const float* __restrict__ bq,
                                const float* __restrict__ bk,
                                const float* __restrict__ bv) {
  int t = blockIdx.x * blockDim.x + threadIdx.x;
  if (t >= B_ * 2 * PAD_ * D_) return;
  int col = t & (D_ - 1);
  int pr  = t >> 9;                 // 0..31
  int b   = pr >> 3;
  int j   = pr & 7;
  int row = b * LP_ + ((j < PAD_) ? j : (LP_ - PAD_ + (j - PAD_)));
  size_t idx = (size_t)row * D_ + col;
  Qp[idx] = bq[col]; Kp[idx] = bk[col]; Vp[idx] = bv[col];
}

// One thread per (b,l,h): 9x9 scores over DK=64, softmax, weighted V sum,
// mean over the 9 query positions -> obar (f16, feeds the Wo GEMM).
__global__ void attn_kernel(const float* __restrict__ Qp,
                            const float* __restrict__ Kp,
                            const float* __restrict__ Vp,
                            _Float16* __restrict__ obar) {
  int t = blockIdx.x * blockDim.x + threadIdx.x;
  if (t >= B_ * L_ * H_) return;
  const int h = t & (H_ - 1);
  const int l = (t >> 3) & (L_ - 1);
  const int b = t >> 14;
  const size_t wbase = ((size_t)(b * LP_ + l)) * D_ + h * DK_;  // window row q = +q*D_
  const float4* Qw = (const float4*)(Qp + wbase);
  const float4* Kw = (const float4*)(Kp + wbase);
  const float4* Vw = (const float4*)(Vp + wbase);
  const int rs4 = D_ / 4;

  float4 acc[16];
#pragma unroll
  for (int i = 0; i < 16; ++i) acc[i] = make_float4(0.f, 0.f, 0.f, 0.f);
  const float scale = 0.125f;  // 1/sqrt(64)

#pragma unroll 1
  for (int q = 0; q < 9; ++q) {
    float s[9];
    float mx = -3.0e38f;
    const float4* qv = Qw + q * rs4;
#pragma unroll
    for (int k = 0; k < 9; ++k) {
      const float4* kv = Kw + k * rs4;
      float dot = 0.f;
#pragma unroll
      for (int d = 0; d < 16; ++d) {
        float4 a = qv[d], c = kv[d];
        dot += a.x * c.x + a.y * c.y + a.z * c.z + a.w * c.w;
      }
      s[k] = dot * scale;
      mx = fmaxf(mx, s[k]);
    }
    float ssum = 0.f;
#pragma unroll
    for (int k = 0; k < 9; ++k) { s[k] = expf(s[k] - mx); ssum += s[k]; }
    float inv = 1.0f / ssum;
#pragma unroll
    for (int k = 0; k < 9; ++k) {
      float wgt = s[k] * inv;
      const float4* vv = Vw + k * rs4;
#pragma unroll
      for (int d = 0; d < 16; ++d) {
        float4 c = vv[d];
        acc[d].x += wgt * c.x; acc[d].y += wgt * c.y;
        acc[d].z += wgt * c.z; acc[d].w += wgt * c.w;
      }
    }
  }
  const float inv9 = 1.0f / 9.0f;
  _Float16* op = obar + ((size_t)(b * L_ + l)) * D_ + h * DK_;
#pragma unroll
  for (int i = 0; i < 16; ++i) {
    op[4 * i + 0] = (_Float16)(acc[i].x * inv9);
    op[4 * i + 1] = (_Float16)(acc[i].y * inv9);
    op[4 * i + 2] = (_Float16)(acc[i].z * inv9);
    op[4 * i + 3] = (_Float16)(acc[i].w * inv9);
  }
}

// LayerNorm over D=512 of (a + c); one wave32 per row, shfl_xor reduction.
// Writes f32 (residual carry / final output) and optionally f16 (GEMM input).
__global__ void add_ln_kernel(const float* __restrict__ a,
                              const float* __restrict__ c,
                              const float* __restrict__ g,
                              const float* __restrict__ be,
                              float* __restrict__ out32,
                              _Float16* __restrict__ out16) {
  int wid  = (blockIdx.x * blockDim.x + threadIdx.x) >> 5;
  int lane = threadIdx.x & 31;
  if (wid >= M_) return;
  const size_t base = (size_t)wid * D_;
  float vals[16];
  float s = 0.f, s2 = 0.f;
#pragma unroll
  for (int i = 0; i < 16; ++i) {
    int col = lane + i * 32;
    float v = a[base + col] + c[base + col];
    vals[i] = v; s += v; s2 += v * v;
  }
#pragma unroll
  for (int off = 16; off; off >>= 1) {
    s  += __shfl_xor(s,  off, 32);
    s2 += __shfl_xor(s2, off, 32);
  }
  float mean = s * (1.0f / D_);
  float var  = s2 * (1.0f / D_) - mean * mean;   // biased, matches jnp.var
  float inv  = rsqrtf(var + 1e-5f);
#pragma unroll
  for (int i = 0; i < 16; ++i) {
    int col = lane + i * 32;
    float y = (vals[i] - mean) * inv * g[col] + be[col];
    if (out32) out32[base + col] = y;
    if (out16) out16[base + col] = (_Float16)y;
  }
}

// ---------------------------------------------------------------------------
extern "C" void kernel_launch(void* const* d_in, const int* in_sizes, int n_in,
                              void* d_out, int out_size, void* d_ws, size_t ws_size,
                              hipStream_t stream) {
  const float* x  = (const float*)d_in[0];
  const float* Wq = (const float*)d_in[1];
  const float* bq = (const float*)d_in[2];
  const float* Wk = (const float*)d_in[3];
  const float* bk = (const float*)d_in[4];
  const float* Wv = (const float*)d_in[5];
  const float* bv = (const float*)d_in[6];
  const float* Wo = (const float*)d_in[7];
  const float* bo = (const float*)d_in[8];
  const float* W1 = (const float*)d_in[9];
  const float* b1 = (const float*)d_in[10];
  const float* W2 = (const float*)d_in[11];
  const float* b2 = (const float*)d_in[12];
  const float* g1 = (const float*)d_in[13];
  const float* be1 = (const float*)d_in[14];
  const float* g2 = (const float*)d_in[15];
  const float* be2 = (const float*)d_in[16];
  float* out = (float*)d_out;

  char* ws = (char*)d_ws;
  size_t off = 0;
  auto alloc = [&](size_t bytes) -> char* {
    char* p = ws + off;
    off += (bytes + 255) & ~(size_t)255;
    return p;
  };
  _Float16* xh    = (_Float16*)alloc((size_t)M_ * D_ * 2);
  _Float16* WtQ   = (_Float16*)alloc((size_t)D_ * D_ * 2);
  _Float16* WtK   = (_Float16*)alloc((size_t)D_ * D_ * 2);
  _Float16* WtV   = (_Float16*)alloc((size_t)D_ * D_ * 2);
  _Float16* WtO   = (_Float16*)alloc((size_t)D_ * D_ * 2);
  _Float16* W1t   = (_Float16*)alloc((size_t)D_ * DF_ * 2);
  _Float16* W2t   = (_Float16*)alloc((size_t)DF_ * D_ * 2);
  float*    Qp    = (float*)alloc((size_t)B_ * LP_ * D_ * 4);
  float*    Kp    = (float*)alloc((size_t)B_ * LP_ * D_ * 4);
  float*    Vp    = (float*)alloc((size_t)B_ * LP_ * D_ * 4);
  _Float16* obar  = (_Float16*)alloc((size_t)M_ * D_ * 2);
  float*    aout  = (float*)alloc((size_t)M_ * D_ * 4);
  float*    x1f   = (float*)alloc((size_t)M_ * D_ * 4);
  _Float16* x1h   = (_Float16*)alloc((size_t)M_ * D_ * 2);
  _Float16* hffn  = (_Float16*)alloc((size_t)M_ * DF_ * 2);
  float*    yf    = (float*)alloc((size_t)M_ * D_ * 4);
  (void)ws_size; (void)in_sizes; (void)n_in; (void)out_size;

  const int TB = 256;
  auto gblk = [](int n) { return (n + 255) / 256; };
  auto gemm_blocks = [](int Mm, int Nn) { return ((Mm / 32) * (Nn / 64) * 32 + 255) / 256; };

  // 1. Stage activations & weights in f16 (weights transposed to N x K).
  convert_f32_to_f16<<<gblk(M_ * D_), TB, 0, stream>>>(x, xh, M_ * D_);
  transpose_to_f16<<<gblk(D_ * D_),  TB, 0, stream>>>(Wq, WtQ, D_, D_);
  transpose_to_f16<<<gblk(D_ * D_),  TB, 0, stream>>>(Wk, WtK, D_, D_);
  transpose_to_f16<<<gblk(D_ * D_),  TB, 0, stream>>>(Wv, WtV, D_, D_);
  transpose_to_f16<<<gblk(D_ * D_),  TB, 0, stream>>>(Wo, WtO, D_, D_);
  transpose_to_f16<<<gblk(D_ * DF_), TB, 0, stream>>>(W1, W1t, D_, DF_);
  transpose_to_f16<<<gblk(DF_ * D_), TB, 0, stream>>>(W2, W2t, DF_, D_);
  pad_init_kernel<<<gblk(B_ * 2 * PAD_ * D_), TB, 0, stream>>>(Qp, Kp, Vp, bq, bk, bv);

  // 2. QKV projections (WMMA), scattered into padded (B, LP, D) f32 layout.
  gemm_wmma_kernel<<<gemm_blocks(M_, D_), TB, 0, stream>>>(xh, WtQ, bq, Qp, nullptr, M_, D_, D_, 1);
  gemm_wmma_kernel<<<gemm_blocks(M_, D_), TB, 0, stream>>>(xh, WtK, bk, Kp, nullptr, M_, D_, D_, 1);
  gemm_wmma_kernel<<<gemm_blocks(M_, D_), TB, 0, stream>>>(xh, WtV, bv, Vp, nullptr, M_, D_, D_, 1);

  // 3. Windowed attention (f32) -> mean over window -> obar (f16).
  attn_kernel<<<gblk(B_ * L_ * H_), TB, 0, stream>>>(Qp, Kp, Vp, obar);

  // 4. Output projection, residual + LN1.
  gemm_wmma_kernel<<<gemm_blocks(M_, D_), TB, 0, stream>>>(obar, WtO, bo, aout, nullptr, M_, D_, D_, 0);
  add_ln_kernel<<<(M_ / 8), TB, 0, stream>>>(x, aout, g1, be1, x1f, x1h);

  // 5. FFN: GELU(x1 @ W1 + b1) @ W2 + b2, residual + LN2 -> out.
  gemm_wmma_kernel<<<gemm_blocks(M_, DF_), TB, 0, stream>>>(x1h, W1t, b1, nullptr, hffn, M_, DF_, D_, 2);
  gemm_wmma_kernel<<<gemm_blocks(M_, D_),  TB, 0, stream>>>(hffn, W2t, b2, yf, nullptr, M_, D_, DF_, 0);
  add_ln_kernel<<<(M_ / 8), TB, 0, stream>>>(x1f, yf, g2, be2, out, nullptr);
}
